// S5Classifier_9354438771046
// MI455X (gfx1250) — matrix operand order repaired and checked by reading, not compiled, not run on hardware
//
#include <hip/hip_runtime.h>
#include <hip/hip_bf16.h>
#include <math.h>

#define H_DIM 256
#define P_DIM 256
#define BATCH_N 8
#define SEQ_N 8192
#define BL (BATCH_N * SEQ_N) /* 65536 */
#define NCLS 5
#define NCHUNK 64
#define CHLEN 128 /* 64*128 == 8192 */
#define LDS_STRIDE 264 /* 256 + 8 pad -> 528B rows, bank-conflict-free */

typedef __bf16 bf16;
typedef __attribute__((ext_vector_type(16))) __bf16 v16bf;
typedef __attribute__((ext_vector_type(8)))  __bf16 v8bf;
typedef __attribute__((ext_vector_type(8)))  float  v8f;
typedef __attribute__((ext_vector_type(4)))  float  v4f;

union AF { v16bf v; v8bf h[2]; bf16 e[16]; };

static __device__ inline v8f wmma_bf16(v16bf a, v16bf b, v8f c) {
  // 8 args: (neg_a, A, neg_b, B, c_mod, C, reuse_a, reuse_b)
  return __builtin_amdgcn_wmma_f32_16x16x32_bf16(false, a, false, b, (short)0, c,
                                                 false, false);
}

static __device__ inline float gelu_erf(float x) {
  return 0.5f * x * (1.0f + erff(x * 0.70710678118654752440f));
}

// Async-stage one 64-row x 256-col bf16 weight panel into padded LDS.
// Panel source is contiguous per row (row stride K elements in global).
// 2048 b128 transfers, 8 per thread, tracked on ASYNCcnt.
template <int K>
static __device__ inline void stage_panel_async(const bf16* __restrict__ wblk,
                                                bf16* __restrict__ sW,
                                                int tid) {
  uint32_t lbase = (uint32_t)(uintptr_t)&sW[0];
#pragma unroll
  for (int i = 0; i < 8; i++) {
    int li = tid + 256 * i;   // 0..2047
    int row = li >> 5;        // 64 rows
    int within = li & 31;     // 32 x 16B per row
    const bf16* gp = wblk + (size_t)row * K + within * 8;
    uint32_t ldst = lbase + (uint32_t)(row * (LDS_STRIDE * 2) + within * 16);
    asm volatile("global_load_async_to_lds_b128 %0, %1, off"
                 :
                 : "v"(ldst), "v"(gp)
                 : "memory");
  }
  asm volatile("s_wait_asynccnt 0x0" ::: "memory");
}

// ---------------------------------------------------------------------------
// Encoder: x[b,l,0]*enc_w[h] + enc_b[h]  (D_IN == 1)
// ---------------------------------------------------------------------------
__global__ __launch_bounds__(256) void k_encoder(const float* __restrict__ Xin,
                                                 const float* __restrict__ Wenc,
                                                 const float* __restrict__ Benc,
                                                 float* __restrict__ Xo) {
  size_t i = (size_t)blockIdx.x * 256 + threadIdx.x;
  size_t bl = i >> 8;
  int hh = (int)(i & 255);
  Xo[i] = Xin[bl] * Wenc[hh] + Benc[hh];
}

// ---------------------------------------------------------------------------
// LayerNorm (optionally two back-to-back LNs). One wave per 256-wide row.
// ---------------------------------------------------------------------------
__global__ __launch_bounds__(256) void k_layernorm2(
    const float* __restrict__ X, const float* __restrict__ w1,
    const float* __restrict__ b1, const float* __restrict__ w2,
    const float* __restrict__ b2, float* __restrict__ Yf,
    bf16* __restrict__ Yb) {
  const int lane = threadIdx.x & 31;
  const int wv = threadIdx.x >> 5;
  const size_t row = (size_t)blockIdx.x * 8 + wv;
  const float* xp = X + row * H_DIM;
  float v[8];
  float s = 0.f, sq = 0.f;
#pragma unroll
  for (int i = 0; i < 8; i++) {
    v[i] = xp[lane + 32 * i];
    s += v[i];
    sq += v[i] * v[i];
  }
#pragma unroll
  for (int o = 16; o >= 1; o >>= 1) {
    s += __shfl_xor(s, o, 32);
    sq += __shfl_xor(sq, o, 32);
  }
  float mu = s * (1.0f / H_DIM);
  float var = sq * (1.0f / H_DIM) - mu * mu;
  float rs = rsqrtf(var + 1e-5f);
#pragma unroll
  for (int i = 0; i < 8; i++) {
    int c = lane + 32 * i;
    v[i] = (v[i] - mu) * rs * w1[c] + b1[c];
  }
  if (w2 != nullptr) {
    s = 0.f; sq = 0.f;
#pragma unroll
    for (int i = 0; i < 8; i++) { s += v[i]; sq += v[i] * v[i]; }
#pragma unroll
    for (int o = 16; o >= 1; o >>= 1) {
      s += __shfl_xor(s, o, 32);
      sq += __shfl_xor(sq, o, 32);
    }
    float mu2 = s * (1.0f / H_DIM);
    float var2 = sq * (1.0f / H_DIM) - mu2 * mu2;
    float rs2 = rsqrtf(var2 + 1e-5f);
#pragma unroll
    for (int i = 0; i < 8; i++) {
      int c = lane + 32 * i;
      v[i] = (v[i] - mu2) * rs2 * w2[c] + b2[c];
    }
  }
#pragma unroll
  for (int i = 0; i < 8; i++) {
    int c = lane + 32 * i;
    Yf[row * H_DIM + c] = v[i];
    Yb[row * H_DIM + c] = (bf16)v[i];
  }
}

// ---------------------------------------------------------------------------
// WMMA GEMM: C[M,N] = A[M,K](bf16) x W[N,K]^T(bf16).
// Block: 8 waves x 64 rows = 512 rows x 64 cols; weight panel staged in LDS
// by async loads. Wave: 4 M-tiles x 4 N-tiles = 16 wmma per 32-wide k-step.
// ---------------------------------------------------------------------------
template <int K, int N, typename OutT>
__global__ __launch_bounds__(256) void k_gemm_AbW(const bf16* __restrict__ A,
                                                  const bf16* __restrict__ W,
                                                  OutT* __restrict__ C) {
  __shared__ bf16 sW[64 * LDS_STRIDE];
  const int tid = threadIdx.x;
  const int lane = tid & 31;
  const int wv = tid >> 5;
  const int half = lane >> 4;
  const int l16 = lane & 15;
  const int nBase = blockIdx.x * 64;
  const int mBase = blockIdx.y * 512 + wv * 64;

  size_t arow[4];
#pragma unroll
  for (int i = 0; i < 4; i++) arow[i] = (size_t)(mBase + i * 16 + l16) * K;

  v8f acc[4][4] = {};
#pragma unroll
  for (int kp = 0; kp < K; kp += 256) {
    __syncthreads();
    stage_panel_async<K>(W + (size_t)nBase * K + kp, sW, tid);
    __syncthreads();
    for (int k2 = 0; k2 < 256; k2 += 32) {
      const int k = kp + k2;
      AF a[4];
#pragma unroll
      for (int i = 0; i < 4; i++) {
        a[i].h[0] = *(const v8bf*)(A + arow[i] + k + half * 8);
        a[i].h[1] = *(const v8bf*)(A + arow[i] + k + 16 + half * 8);
        __builtin_prefetch(A + arow[i] + k + 128, 0, 3);
      }
      AF b[4];
#pragma unroll
      for (int t = 0; t < 4; t++) {
        const bf16* wp = &sW[(t * 16 + l16) * LDS_STRIDE + k2 + half * 8];
        b[t].h[0] = *(const v8bf*)wp;
        b[t].h[1] = *(const v8bf*)(wp + 16);
      }
#pragma unroll
      for (int t = 0; t < 4; t++)
#pragma unroll
        for (int i = 0; i < 4; i++)
          acc[i][t] = wmma_bf16(a[i].v, b[t].v, acc[i][t]);
    }
  }
#pragma unroll
  for (int i = 0; i < 4; i++)
#pragma unroll
    for (int t = 0; t < 4; t++) {
      const int col = nBase + t * 16 + l16;
      const int row0 = mBase + i * 16 + half * 8;
#pragma unroll
      for (int j = 0; j < 8; j++)
        C[(size_t)(row0 + j) * N + col] = (OutT)acc[i][t][j];
    }
}

// ---------------------------------------------------------------------------
// Readout GEMM: A = xs f32 [M,512] (re|im) converted to bf16 on the fly,
// W = [256][512] bf16 holding [2*C_re | -2*C_im].
// Epilogue: y += D[n]*fx;  x2 = gelu(y) + fx.
// ---------------------------------------------------------------------------
__global__ __launch_bounds__(256) void k_gemm_readout(
    const float* __restrict__ A, const bf16* __restrict__ W,
    const float* __restrict__ FX, const float* __restrict__ Dv,
    float* __restrict__ X2) {
  constexpr int K = 512;
  __shared__ bf16 sW[64 * LDS_STRIDE];
  const int tid = threadIdx.x;
  const int lane = tid & 31;
  const int wv = tid >> 5;
  const int half = lane >> 4;
  const int l16 = lane & 15;
  const int nBase = blockIdx.x * 64;
  const int mBase = blockIdx.y * 512 + wv * 64;

  size_t arow[4];
#pragma unroll
  for (int i = 0; i < 4; i++) arow[i] = (size_t)(mBase + i * 16 + l16) * K;

  v8f acc[4][4] = {};
#pragma unroll
  for (int kp = 0; kp < K; kp += 256) {
    __syncthreads();
    stage_panel_async<K>(W + (size_t)nBase * K + kp, sW, tid);
    __syncthreads();
    for (int k2 = 0; k2 < 256; k2 += 32) {
      const int k = kp + k2;
      AF a[4];
#pragma unroll
      for (int i = 0; i < 4; i++) {
        const float* ap = A + arow[i] + k + half * 8;
        v4f x0 = *(const v4f*)ap;
        v4f x1 = *(const v4f*)(ap + 4);
        v4f x2 = *(const v4f*)(ap + 16);
        v4f x3 = *(const v4f*)(ap + 20);
#pragma unroll
        for (int e = 0; e < 4; e++) {
          a[i].e[e] = (bf16)x0[e];
          a[i].e[4 + e] = (bf16)x1[e];
          a[i].e[8 + e] = (bf16)x2[e];
          a[i].e[12 + e] = (bf16)x3[e];
        }
        __builtin_prefetch(ap + 128, 0, 3);
      }
      AF b[4];
#pragma unroll
      for (int t = 0; t < 4; t++) {
        const bf16* wp = &sW[(t * 16 + l16) * LDS_STRIDE + k2 + half * 8];
        b[t].h[0] = *(const v8bf*)wp;
        b[t].h[1] = *(const v8bf*)(wp + 16);
      }
#pragma unroll
      for (int t = 0; t < 4; t++)
#pragma unroll
        for (int i = 0; i < 4; i++)
          acc[i][t] = wmma_bf16(a[i].v, b[t].v, acc[i][t]);
    }
  }
#pragma unroll
  for (int i = 0; i < 4; i++)
#pragma unroll
    for (int t = 0; t < 4; t++) {
      const int col = nBase + t * 16 + l16;
      const int row0 = mBase + i * 16 + half * 8;
      const float dv = Dv[col];
#pragma unroll
      for (int j = 0; j < 8; j++) {
        size_t o = (size_t)(row0 + j) * H_DIM + col;
        float fx = FX[o];
        float y = acc[i][t][j] + fx * dv;
        X2[o] = gelu_erf(y) + fx;
      }
    }
}

// ---------------------------------------------------------------------------
// FFN dec GEMM: A = a*gelu(g) bf16 [M,256], W = ff_dec [256][256] bf16.
// Epilogue: x_new = x_old + (acc + fx2)   (in-place update of residual x).
// ---------------------------------------------------------------------------
__global__ __launch_bounds__(256) void k_gemm_dec(const bf16* __restrict__ A,
                                                  const bf16* __restrict__ W,
                                                  const float* __restrict__ FX2,
                                                  float* __restrict__ X) {
  constexpr int K = 256;
  __shared__ bf16 sW[64 * LDS_STRIDE];
  const int tid = threadIdx.x;
  const int lane = tid & 31;
  const int wv = tid >> 5;
  const int half = lane >> 4;
  const int l16 = lane & 15;
  const int nBase = blockIdx.x * 64;
  const int mBase = blockIdx.y * 512 + wv * 64;

  size_t arow[4];
#pragma unroll
  for (int i = 0; i < 4; i++) arow[i] = (size_t)(mBase + i * 16 + l16) * K;

  v8f acc[4][4] = {};
  stage_panel_async<K>(W + (size_t)nBase * K, sW, tid);
  __syncthreads();
  for (int k = 0; k < K; k += 32) {
    AF a[4];
#pragma unroll
    for (int i = 0; i < 4; i++) {
      a[i].h[0] = *(const v8bf*)(A + arow[i] + k + half * 8);
      a[i].h[1] = *(const v8bf*)(A + arow[i] + k + 16 + half * 8);
      __builtin_prefetch(A + arow[i] + k + 128, 0, 3);
    }
    AF b[4];
#pragma unroll
    for (int t = 0; t < 4; t++) {
      const bf16* wp = &sW[(t * 16 + l16) * LDS_STRIDE + k + half * 8];
      b[t].h[0] = *(const v8bf*)wp;
      b[t].h[1] = *(const v8bf*)(wp + 16);
    }
#pragma unroll
    for (int t = 0; t < 4; t++)
#pragma unroll
      for (int i = 0; i < 4; i++)
        acc[i][t] = wmma_bf16(a[i].v, b[t].v, acc[i][t]);
  }
#pragma unroll
  for (int i = 0; i < 4; i++)
#pragma unroll
    for (int t = 0; t < 4; t++) {
      const int col = nBase + t * 16 + l16;
      const int row0 = mBase + i * 16 + half * 8;
#pragma unroll
      for (int j = 0; j < 8; j++) {
        size_t o = (size_t)(row0 + j) * H_DIM + col;
        X[o] = X[o] + acc[i][t][j] + FX2[o];
      }
    }
}

// ---------------------------------------------------------------------------
// GEGLU: out[m,c] = h[m,c] * gelu(h[m,256+c])
// ---------------------------------------------------------------------------
__global__ __launch_bounds__(256) void k_geglu(const bf16* __restrict__ Hh,
                                               bf16* __restrict__ O) {
  size_t i = (size_t)blockIdx.x * 256 + threadIdx.x;
  size_t row = i >> 8;
  int c = (int)(i & 255);
  float a = (float)Hh[row * 512 + c];
  float g = (float)Hh[row * 512 + 256 + c];
  O[i] = (bf16)(a * gelu_erf(g));
}

// ---------------------------------------------------------------------------
// SSM parameter prep. lamb layout: [0:256)=lam_bar_re [256:512)=lam_bar_im
// [512:768)=coef_re [768:1024)=coef_im  where coef = (lam_bar-1)/lam.
// ---------------------------------------------------------------------------
__global__ void k_prep_lam(const float* __restrict__ lam_re,
                           const float* __restrict__ lam_im,
                           const float* __restrict__ log_step,
                           float* __restrict__ lamb) {
  int p = threadIdx.x;
  float dt = expf(log_step[p]);
  float ar = lam_re[p] * dt, ai = lam_im[p] * dt;
  float e = expf(ar);
  float lbr = e * cosf(ai), lbi = e * sinf(ai);
  float nr = lbr - 1.0f, ni = lbi;
  float dr = lam_re[p], di = lam_im[p];
  float inv = 1.0f / (dr * dr + di * di);
  lamb[p] = lbr;
  lamb[256 + p] = lbi;
  lamb[512 + p] = (nr * dr + ni * di) * inv;
  lamb[768 + p] = (ni * dr - nr * di) * inv;
}

// Wb[p][h]=Re(B_bar), Wb[256+p][h]=Im(B_bar); B_bar = coef * (B_re + i B_im)
__global__ void k_prep_wb(const float* __restrict__ Bre,
                          const float* __restrict__ Bim,
                          const float* __restrict__ lamb,
                          bf16* __restrict__ Wb) {
  int h = threadIdx.x;
  int row = blockIdx.x; // 0..511
  int p = row & 255;
  float cr = lamb[512 + p], ci = lamb[768 + p];
  float br = Bre[p * H_DIM + h], bi = Bim[p * H_DIM + h];
  float v = (row < 256) ? (cr * br - ci * bi) : (cr * bi + ci * br);
  Wb[(size_t)row * H_DIM + h] = (bf16)v;
}

// Wc[h][k] = 2*C_re[h,k] (k<256) | -2*C_im[h,k-256]
__global__ void k_prep_wc(const float* __restrict__ Cre,
                          const float* __restrict__ Cim,
                          bf16* __restrict__ Wc) {
  int k = threadIdx.x;  // 0..511
  int h = blockIdx.x;   // 0..255
  float v = (k < 256) ? (2.0f * Cre[h * 256 + k]) : (-2.0f * Cim[h * 256 + (k - 256)]);
  Wc[(size_t)h * 512 + k] = (bf16)v;
}

__global__ void k_cvt_bf16(const float* __restrict__ S, bf16* __restrict__ D,
                           int n) {
  int i = blockIdx.x * 256 + threadIdx.x;
  if (i < n) D[i] = (bf16)S[i];
}

// ---------------------------------------------------------------------------
// Chunked complex scan: s_l = lam*s_{l-1} + Bu_l, in-place on BU [BL][512]
// (cols 0..255 = re, 256..511 = im). 64 chunks of 128.
// ---------------------------------------------------------------------------
__global__ __launch_bounds__(256) void k_scan_chunk(float* __restrict__ BU,
                                                    const float* __restrict__ lamb,
                                                    float* __restrict__ chst) {
  int p = threadIdx.x, c = blockIdx.x, b = blockIdx.y;
  float lr = lamb[p], li = lamb[256 + p];
  float sr = 0.f, si = 0.f;
  size_t base = (((size_t)b * SEQ_N) + (size_t)c * CHLEN) * 512 + p;
  for (int l = 0; l < CHLEN; l++) {
    size_t idx = base + (size_t)l * 512;
    float ur = BU[idx], ui = BU[idx + 256];
    float nr = lr * sr - li * si + ur;
    float ni = lr * si + li * sr + ui;
    sr = nr; si = ni;
    BU[idx] = sr;
    BU[idx + 256] = si;
  }
  size_t ci = ((size_t)b * NCHUNK + c) * 512 + p;
  chst[ci] = sr;
  chst[ci + 256] = si;
}

__global__ void k_scan_combine(const float* __restrict__ chst,
                               const float* __restrict__ lamb,
                               float* __restrict__ chpref) {
  int p = threadIdx.x, b = blockIdx.x;
  float lr = lamb[p], li = lamb[256 + p];
  float ar = lr, ai = li; // lam^128 by 7 squarings
#pragma unroll
  for (int i = 0; i < 7; i++) {
    float nr = ar * ar - ai * ai, ni = 2.f * ar * ai;
    ar = nr; ai = ni;
  }
  float sr = 0.f, si = 0.f;
  for (int c = 0; c < NCHUNK; c++) {
    size_t ci = ((size_t)b * NCHUNK + c) * 512 + p;
    chpref[ci] = sr;
    chpref[ci + 256] = si;
    float er = chst[ci], ei = chst[ci + 256];
    float nr = ar * sr - ai * si + er;
    float ni = ar * si + ai * sr + ei;
    sr = nr; si = ni;
  }
}

__global__ __launch_bounds__(256) void k_scan_fix(float* __restrict__ BU,
                                                  const float* __restrict__ lamb,
                                                  const float* __restrict__ chpref) {
  int p = threadIdx.x, c = blockIdx.x, b = blockIdx.y;
  if (c == 0) return;
  float lr = lamb[p], li = lamb[256 + p];
  size_t ci = ((size_t)b * NCHUNK + c) * 512 + p;
  float sr = chpref[ci], si = chpref[ci + 256];
  float tr = lr * sr - li * si; // lam^1 * S
  float ti = lr * si + li * sr;
  size_t base = (((size_t)b * SEQ_N) + (size_t)c * CHLEN) * 512 + p;
  for (int l = 0; l < CHLEN; l++) {
    size_t idx = base + (size_t)l * 512;
    BU[idx] += tr;
    BU[idx + 256] += ti;
    float nr = lr * tr - li * ti, ni = lr * ti + li * tr;
    tr = nr; ti = ni;
  }
}

// ---------------------------------------------------------------------------
// Mean pool over sequence, then classifier head.
// ---------------------------------------------------------------------------
__global__ __launch_bounds__(256) void k_pool(const float* __restrict__ X,
                                              float* __restrict__ XM) {
  int hh = threadIdx.x, b = blockIdx.x;
  float s = 0.f;
  size_t base = (size_t)b * SEQ_N * H_DIM + hh;
  for (int l = 0; l < SEQ_N; l++) s += X[base + (size_t)l * H_DIM];
  XM[b * H_DIM + hh] = s * (1.0f / SEQ_N);
}

__global__ void k_head(const float* __restrict__ XM,
                       const float* __restrict__ Wh,
                       const float* __restrict__ Bh, float* __restrict__ Out) {
  int t = threadIdx.x;
  if (t >= BATCH_N * NCLS) return;
  int b = t / NCLS, c = t % NCLS;
  float s = Bh[c];
  for (int hh = 0; hh < H_DIM; hh++) s += XM[b * H_DIM + hh] * Wh[c * H_DIM + hh];
  Out[t] = s;
}

// ---------------------------------------------------------------------------
// Host launcher
// ---------------------------------------------------------------------------
extern "C" void kernel_launch(void* const* d_in, const int* in_sizes, int n_in,
                              void* d_out, int out_size, void* d_ws,
                              size_t ws_size, hipStream_t stream) {
  (void)in_sizes; (void)n_in; (void)out_size; (void)ws_size;
  const float* Xin = (const float*)d_in[0];
  const float* EncW = (const float*)d_in[1];
  const float* EncB = (const float*)d_in[2];
  const float* HeadW = (const float*)d_in[3];
  const float* HeadB = (const float*)d_in[4];

  char* ws = (char*)d_ws;
  const size_t SZ_ACT = (size_t)BL * H_DIM * sizeof(float); // 64 MB
  size_t off = 0;
  float* buf_x = (float*)(ws + off); off += SZ_ACT;
  float* buf_fx = (float*)(ws + off); off += SZ_ACT;
  bf16* buf_fxbf = (bf16*)(ws + off); off += (size_t)BL * H_DIM * sizeof(bf16);
  float* buf_bu = (float*)(ws + off); off += (size_t)BL * 512 * sizeof(float);
  bf16* buf_h = (bf16*)buf_bu; // FFN hidden aliases the scan buffer
  float* buf_x2 = (float*)(ws + off); off += SZ_ACT;
  float* buf_lam = (float*)(ws + off); off += 1024 * sizeof(float);
  float* buf_chst = (float*)(ws + off); off += (size_t)BATCH_N * NCHUNK * 512 * sizeof(float);
  float* buf_chpref = (float*)(ws + off); off += (size_t)BATCH_N * NCHUNK * 512 * sizeof(float);
  bf16* buf_wb = (bf16*)(ws + off); off += (size_t)512 * 256 * sizeof(bf16);
  bf16* buf_wc = (bf16*)(ws + off); off += (size_t)256 * 512 * sizeof(bf16);
  bf16* buf_wenc = (bf16*)(ws + off); off += (size_t)512 * 256 * sizeof(bf16);
  bf16* buf_wdec = (bf16*)(ws + off); off += (size_t)256 * 256 * sizeof(bf16);
  float* buf_xm = (float*)(ws + off); off += (size_t)BATCH_N * H_DIM * sizeof(float);

  // 1. Encoder
  k_encoder<<<BL * H_DIM / 256, 256, 0, stream>>>(Xin, EncW, EncB, buf_x);

  for (int l = 0; l < 3; l++) {
    const int base = 5 + 16 * l;
    const float* ln_w = (const float*)d_in[base + 0];
    const float* ln_b = (const float*)d_in[base + 1];
    const float* attn_w = (const float*)d_in[base + 2];
    const float* attn_b = (const float*)d_in[base + 3];
    const float* lam_re = (const float*)d_in[base + 4];
    const float* lam_im = (const float*)d_in[base + 5];
    const float* B_re = (const float*)d_in[base + 6];
    const float* B_im = (const float*)d_in[base + 7];
    const float* C_re = (const float*)d_in[base + 8];
    const float* C_im = (const float*)d_in[base + 9];
    const float* Dv = (const float*)d_in[base + 10];
    const float* log_step = (const float*)d_in[base + 11];
    const float* ffn_w = (const float*)d_in[base + 12];
    const float* ffn_b = (const float*)d_in[base + 13];
    const float* ff_enc = (const float*)d_in[base + 14];
    const float* ff_dec = (const float*)d_in[base + 15];

    // parameter prep (tiny)
    k_prep_lam<<<1, 256, 0, stream>>>(lam_re, lam_im, log_step, buf_lam);
    k_prep_wb<<<512, 256, 0, stream>>>(B_re, B_im, buf_lam, buf_wb);
    k_prep_wc<<<256, 512, 0, stream>>>(C_re, C_im, buf_wc);
    k_cvt_bf16<<<512, 256, 0, stream>>>(ff_enc, buf_wenc, 512 * 256);
    k_cvt_bf16<<<256, 256, 0, stream>>>(ff_dec, buf_wdec, 256 * 256);

    // fx = LN_attn(LN_pre(x))  (f32 + bf16)
    k_layernorm2<<<BL / 8, 256, 0, stream>>>(buf_x, ln_w, ln_b, attn_w, attn_b,
                                             buf_fx, buf_fxbf);
    // Bu = fx @ B_bar^T  (re|im): [65536,256] x [256,512]
    k_gemm_AbW<256, 512, float><<<dim3(8, 128), 256, 0, stream>>>(buf_fxbf,
                                                                  buf_wb, buf_bu);
    // complex scan along L
    k_scan_chunk<<<dim3(NCHUNK, BATCH_N), 256, 0, stream>>>(buf_bu, buf_lam,
                                                            buf_chst);
    k_scan_combine<<<BATCH_N, 256, 0, stream>>>(buf_chst, buf_lam, buf_chpref);
    k_scan_fix<<<dim3(NCHUNK, BATCH_N), 256, 0, stream>>>(buf_bu, buf_lam,
                                                          buf_chpref);
    // y = 2Re(C xs) + D*fx; x2 = gelu(y) + fx
    k_gemm_readout<<<dim3(4, 128), 256, 0, stream>>>(buf_bu, buf_wc, buf_fx, Dv,
                                                     buf_x2);
    // fx2 = LN_ff(x2)
    k_layernorm2<<<BL / 8, 256, 0, stream>>>(buf_x2, ffn_w, ffn_b, nullptr,
                                             nullptr, buf_fx, buf_fxbf);
    // h = fx2 @ ff_enc^T : [65536,256] x [256,512] -> bf16
    k_gemm_AbW<256, 512, bf16><<<dim3(8, 128), 256, 0, stream>>>(buf_fxbf,
                                                                 buf_wenc, buf_h);
    // ag = a * gelu(g)  (reuses fx2 bf16 buffer)
    k_geglu<<<BL * H_DIM / 256, 256, 0, stream>>>(buf_h, buf_fxbf);
    // x = x + (ag @ ff_dec^T + fx2)
    k_gemm_dec<<<dim3(4, 128), 256, 0, stream>>>(buf_fxbf, buf_wdec, buf_fx,
                                                 buf_x);
  }

  k_pool<<<BATCH_N, 256, 0, stream>>>(buf_x, buf_xm);
  k_head<<<1, 64, 0, stream>>>(buf_xm, HeadW, HeadB, (float*)d_out);
}
